// TensorfModel_14482629722461
// MI455X (gfx1250) — matrix operand order repaired and checked by reading, not compiled, not run on hardware
//
#include <hip/hip_runtime.h>
#include <hip/hip_bf16.h>

#define GRID_N 300
#define RC 48
#define RD 16
#define FEAT 27
#define HID 128
#define DIN 150      // (1+2*PE)*(3+FEAT)
#define DINP 160     // padded to multiple of 32 for K of wmma f16
#define TILE_PTS 128
#define THREADS 128
#define NPTS (4096 * 128)

typedef __attribute__((ext_vector_type(16))) _Float16 v16h;
typedef __attribute__((ext_vector_type(8)))  float    v8f;

union AFrag { unsigned int u[8]; v16h h; };
union BFrag { uint4 q[2];        v16h h; };

// ---------------------------------------------------------------------------
// A-fragment (16x32 f16, MxK) gather from a row-major LDS matrix (stride DINP).
// Per CDNA5 ISA: lanes 0-15 hold M=lane, K={0..7,16..23}; lanes 16-31 hold
// M=lane-16, K={8..15,24..31}. K pairs are contiguous -> 8 x 32-bit LDS loads.
// ---------------------------------------------------------------------------
__device__ __forceinline__ v16h load_a_frag(const _Float16* sM, int mtile,
                                            int kofs, int lane) {
  const _Float16* row = sM + (mtile * 16 + (lane & 15)) * DINP + kofs + (lane >> 4) * 8;
  AFrag f;
#pragma unroll
  for (int v = 0; v < 8; ++v) {
    int kb = 2 * v + (v < 4 ? 0 : 8);
    f.u[v] = *(const unsigned int*)(row + kb);
  }
  return f.h;
}

// B fragments are pre-swizzled by the prep kernel into [frag][lane][16 halfs]
// order -> each lane does two contiguous b128 loads.
__device__ __forceinline__ v16h load_b_frag(const _Float16* g, int fragIdx, int lane) {
  const uint4* p = (const uint4*)(g + ((size_t)fragIdx * 32 + lane) * 16);
  BFrag f;
  f.q[0] = p[0];
  f.q[1] = p[1];
  return f.h;
}

// ---------------------------------------------------------------------------
// Prep: convert w1/w2 to f16 in WMMA B-matrix fragment order, and hoist the
// degenerate vector-sample constants (reference samples vectors at y=149.5
// always: 0.5*(v[149]+v[150]) per channel).
// B layout assumed (K x N, 16-bit): lane n = L%16; lanes 0-15 carry K={0..7,16..23},
// lanes 16-31 carry K={8..15,24..31} of the 32-K step, 2 halfs per VGPR.
// ---------------------------------------------------------------------------
__global__ void prep_kernel(const float* __restrict__ w1, const float* __restrict__ w2,
                            const float* __restrict__ vmc_vec, const float* __restrict__ vmd_vec,
                            _Float16* __restrict__ w1h, _Float16* __restrict__ w2h,
                            float* __restrict__ vcc, float* __restrict__ vcd) {
  int idx = blockIdx.x * blockDim.x + threadIdx.x;
  if (idx < 8 * 5 * 512) {               // w1: 8 ntiles x 5 ksteps x 32 lanes x 16 halfs
    int j = idx & 15, lane = (idx >> 4) & 31, rem = idx >> 9;
    int ntile = rem / 5, ks = rem % 5;
    int n = ntile * 16 + (lane & 15);
    int k = ks * 32 + (lane >> 4) * 8 + (j < 8 ? j : 8 + j);
    float v = (k < DIN) ? w1[n * DIN + k] : 0.f;
    w1h[idx] = (_Float16)v;
  }
  if (idx < 8 * 4 * 512) {               // w2: 8 ntiles x 4 ksteps
    int j = idx & 15, lane = (idx >> 4) & 31, rem = idx >> 9;
    int ntile = rem >> 2, ks = rem & 3;
    int n = ntile * 16 + (lane & 15);
    int k = ks * 32 + (lane >> 4) * 8 + (j < 8 ? j : 8 + j);
    w2h[idx] = (_Float16)w2[n * HID + k];
  }
  if (idx < 3 * RC) vcc[idx] = 0.5f * (vmc_vec[idx * GRID_N + 149] + vmc_vec[idx * GRID_N + 150]);
  if (idx < 3 * RD) vcd[idx] = 0.5f * (vmd_vec[idx * GRID_N + 149] + vmd_vec[idx * GRID_N + 150]);
}

// ---------------------------------------------------------------------------
// Fused kernel: 128-thread block == one 128-point tile.
//  phase1: per-thread bilinear gathers (L2-resident planes), feat_w fold,
//          PE sin/cos, build f16 A-matrix (128 x 160) in LDS.
//  phase2: layer1 via v_wmma_f32_16x16x32_f16 (5 K-steps), bias+relu -> LDS.
//  phase3: layer2 via WMMA (4 K-steps), bias+relu -> LDS.
//  phase4: skinny 3x128 output layer + sigmoid + density in VALU.
// ---------------------------------------------------------------------------
__global__ __launch_bounds__(THREADS) void nerf_fused(
    const float* __restrict__ x,
    const float* __restrict__ vmc_mat,
    const float* __restrict__ vmd_mat,
    const float* __restrict__ feat_w,
    const float* __restrict__ b1,
    const float* __restrict__ b2,
    const float* __restrict__ w3,
    const float* __restrict__ b3,
    const _Float16* __restrict__ w1h,
    const _Float16* __restrict__ w2h,
    const float* __restrict__ vcc,
    const float* __restrict__ vcd,
    float* __restrict__ out) {
  __shared__ _Float16 sA[TILE_PTS * DINP];        // 40 KB activation ping-pong
  __shared__ float sFeatW[FEAT * 3 * RC];         // 15.2 KB
  __shared__ float sVcc[3 * RC];
  __shared__ float sVcd[3 * RD];

  const int tid = threadIdx.x;
  const int lane = tid & 31;
  const int wave = tid >> 5;

  for (int i = tid; i < FEAT * 3 * RC; i += THREADS) sFeatW[i] = feat_w[i];
  for (int i = tid; i < 3 * RC; i += THREADS) sVcc[i] = vcc[i];
  for (int i = tid; i < 3 * RD; i += THREADS) sVcd[i] = vcd[i];
  __syncthreads();

  const int p = blockIdx.x * TILE_PTS + tid;
  const float* xp = x + (size_t)p * 6;
  float pos[3] = {xp[0], xp[1], xp[2]};
  float dir[3] = {xp[3], xp[4], xp[5]};

  float feat[FEAT];
#pragma unroll
  for (int f = 0; f < FEAT; ++f) feat[f] = 0.f;
  float density = 0.f;

#pragma unroll
  for (int a = 0; a < 3; ++a) {
    float xg = (pos[(a + 1) % 3] + 1.f) * 0.5f * (float)(GRID_N - 1);
    float yg = (pos[(a + 2) % 3] + 1.f) * 0.5f * (float)(GRID_N - 1);
    float x0f = floorf(xg), y0f = floorf(yg);
    int x0 = (int)fminf(fmaxf(x0f, 0.f), (float)(GRID_N - 1));
    int y0 = (int)fminf(fmaxf(y0f, 0.f), (float)(GRID_N - 1));
    int x1 = x0 + 1; if (x1 > GRID_N - 1) x1 = GRID_N - 1;
    int y1 = y0 + 1; if (y1 > GRID_N - 1) y1 = GRID_N - 1;
    float wx = xg - x0f, wy = yg - y0f;
    float w00 = (1.f - wy) * (1.f - wx), w01 = (1.f - wy) * wx;
    float w10 = wy * (1.f - wx),         w11 = wy * wx;
    int o00 = y0 * GRID_N + x0, o01 = y0 * GRID_N + x1;
    int o10 = y1 * GRID_N + x0, o11 = y1 * GRID_N + x1;

    const float* planeC = vmc_mat + (size_t)a * RC * GRID_N * GRID_N;
    for (int c = 0; c < RC; ++c) {
      const float* pm = planeC + (size_t)c * GRID_N * GRID_N;
      float m = pm[o00] * w00 + pm[o01] * w01 + pm[o10] * w10 + pm[o11] * w11;
      float vmv = m * sVcc[a * RC + c];
      const float* fw = sFeatW + a * RC + c;
#pragma unroll
      for (int f = 0; f < FEAT; ++f) feat[f] = fmaf(vmv, fw[f * (3 * RC)], feat[f]);
    }
    const float* planeD = vmd_mat + (size_t)a * RD * GRID_N * GRID_N;
    for (int c = 0; c < RD; ++c) {
      const float* pm = planeD + (size_t)c * GRID_N * GRID_N;
      float m = pm[o00] * w00 + pm[o01] * w01 + pm[o10] * w10 + pm[o11] * w11;
      density = fmaf(m, sVcd[a * RD + c], density);
    }
  }

  // mlp_in row: [feat 0..26][dir 27..29][pe(feat) 30..137][pe(dir) 138..149][pad]
  {
    _Float16* row = sA + tid * DINP;
    const float PI1 = 3.14159265358979f;
#pragma unroll
    for (int f = 0; f < FEAT; ++f) row[f] = (_Float16)feat[f];
#pragma unroll
    for (int d = 0; d < 3; ++d) row[FEAT + d] = (_Float16)dir[d];
#pragma unroll
    for (int ci = 0; ci < 2; ++ci) {
      float cc = PI1 * (float)(1 << ci);
      int bf = FEAT + 3 + ci * (2 * FEAT);
#pragma unroll
      for (int f = 0; f < FEAT; ++f) {
        float v = cc * feat[f];
        row[bf + f]        = (_Float16)__sinf(v);
        row[bf + FEAT + f] = (_Float16)__cosf(v);
      }
      int bd = FEAT + 3 + 4 * FEAT + ci * 6;
#pragma unroll
      for (int d = 0; d < 3; ++d) {
        float v = cc * dir[d];
        row[bd + d]     = (_Float16)__sinf(v);
        row[bd + 3 + d] = (_Float16)__cosf(v);
      }
    }
#pragma unroll
    for (int i = DIN; i < DINP; ++i) row[i] = (_Float16)0.f;
  }
  __syncthreads();

  const v8f vzero = {0.f, 0.f, 0.f, 0.f, 0.f, 0.f, 0.f, 0.f};

  // ---------------- layer 1: [128x160] x [160x128] ----------------
  {
    v8f acc[2][8];
#pragma unroll
    for (int mt = 0; mt < 2; ++mt)
#pragma unroll
      for (int nt = 0; nt < 8; ++nt) acc[mt][nt] = vzero;

#pragma unroll
    for (int mt = 0; mt < 2; ++mt) {
      int mtile = wave * 2 + mt;
#pragma unroll
      for (int ks = 0; ks < 5; ++ks) {
        v16h afrag = load_a_frag(sA, mtile, ks * 32, lane);
#pragma unroll
        for (int nt = 0; nt < 8; ++nt) {
          v16h bfrag = load_b_frag(w1h, nt * 5 + ks, lane);
          acc[mt][nt] = __builtin_amdgcn_wmma_f32_16x16x32_f16(
              false, afrag, false, bfrag, (short)0, acc[mt][nt], false, false);
        }
      }
    }
    __syncthreads();   // everyone done reading sA before overwrite
    // C layout: lanes 0-15: N=lane, M=vgpr; lanes 16-31: N=lane-16, M=8+vgpr
#pragma unroll
    for (int mt = 0; mt < 2; ++mt) {
      int mrow = (wave * 2 + mt) * 16 + (lane >> 4) * 8;
#pragma unroll
      for (int nt = 0; nt < 8; ++nt) {
        int n = nt * 16 + (lane & 15);
        float bias = b1[n];
#pragma unroll
        for (int r = 0; r < 8; ++r) {
          float hv = fmaxf(acc[mt][nt][r] + bias, 0.f);
          sA[(mrow + r) * DINP + n] = (_Float16)hv;
        }
      }
    }
  }
  __syncthreads();

  // ---------------- layer 2: [128x128] x [128x128] ----------------
  {
    v8f acc[2][8];
#pragma unroll
    for (int mt = 0; mt < 2; ++mt)
#pragma unroll
      for (int nt = 0; nt < 8; ++nt) acc[mt][nt] = vzero;

#pragma unroll
    for (int mt = 0; mt < 2; ++mt) {
      int mtile = wave * 2 + mt;
#pragma unroll
      for (int ks = 0; ks < 4; ++ks) {
        v16h afrag = load_a_frag(sA, mtile, ks * 32, lane);
#pragma unroll
        for (int nt = 0; nt < 8; ++nt) {
          v16h bfrag = load_b_frag(w2h, nt * 4 + ks, lane);
          acc[mt][nt] = __builtin_amdgcn_wmma_f32_16x16x32_f16(
              false, afrag, false, bfrag, (short)0, acc[mt][nt], false, false);
        }
      }
    }
    __syncthreads();
#pragma unroll
    for (int mt = 0; mt < 2; ++mt) {
      int mrow = (wave * 2 + mt) * 16 + (lane >> 4) * 8;
#pragma unroll
      for (int nt = 0; nt < 8; ++nt) {
        int n = nt * 16 + (lane & 15);
        float bias = b2[n];
#pragma unroll
        for (int r = 0; r < 8; ++r) {
          float hv = fmaxf(acc[mt][nt][r] + bias, 0.f);
          sA[(mrow + r) * DINP + n] = (_Float16)hv;
        }
      }
    }
  }
  __syncthreads();

  // ---------------- layer 3 (3x128) + sigmoid + density ----------------
  {
    float c0 = b3[0], c1 = b3[1], c2 = b3[2];
    const _Float16* hrow = sA + tid * DINP;
    for (int n = 0; n < HID; ++n) {
      float hv = (float)hrow[n];
      c0 = fmaf(hv, w3[n], c0);
      c1 = fmaf(hv, w3[HID + n], c1);
      c2 = fmaf(hv, w3[2 * HID + n], c2);
    }
    float4 o;
    o.x = 1.f / (1.f + __expf(-c0));
    o.y = 1.f / (1.f + __expf(-c1));
    o.z = 1.f / (1.f + __expf(-c2));
    o.w = fmaxf(density, 0.f);
    *(float4*)(out + (size_t)p * 4) = o;
  }
}

extern "C" void kernel_launch(void* const* d_in, const int* in_sizes, int n_in,
                              void* d_out, int out_size, void* d_ws, size_t ws_size,
                              hipStream_t stream) {
  (void)in_sizes; (void)n_in; (void)out_size; (void)ws_size;
  const float* x       = (const float*)d_in[0];
  const float* vmc_vec = (const float*)d_in[1];
  const float* vmc_mat = (const float*)d_in[2];
  const float* vmd_vec = (const float*)d_in[3];
  const float* vmd_mat = (const float*)d_in[4];
  const float* feat_w  = (const float*)d_in[5];
  const float* w1      = (const float*)d_in[6];
  const float* b1      = (const float*)d_in[7];
  const float* w2      = (const float*)d_in[8];
  const float* b2      = (const float*)d_in[9];
  const float* w3      = (const float*)d_in[10];
  const float* b3      = (const float*)d_in[11];
  float* out = (float*)d_out;

  char* ws = (char*)d_ws;
  _Float16* w1h = (_Float16*)(ws);                     // 8*5*512 halfs = 40960 B
  _Float16* w2h = (_Float16*)(ws + 40960);             // 8*4*512 halfs = 32768 B
  float*    vcc = (float*)(ws + 40960 + 32768);        // 144 floats
  float*    vcd = (float*)(ws + 40960 + 32768 + 1024); // 48 floats

  prep_kernel<<<(20480 + 255) / 256, 256, 0, stream>>>(w1, w2, vmc_vec, vmd_vec,
                                                       w1h, w2h, vcc, vcd);
  nerf_fused<<<NPTS / TILE_PTS, THREADS, 0, stream>>>(
      x, vmc_mat, vmd_mat, feat_w, b1, b2, w3, b3, w1h, w2h, vcc, vcd, out);
}
